// GraphEncoder_3693671874956
// MI455X (gfx1250) — compile-verified, matrix operand order checked by reading, not run on hardware
//
#include <hip/hip_runtime.h>
#include <stdint.h>

typedef __attribute__((ext_vector_type(16))) __bf16 v16bf;
typedef __attribute__((ext_vector_type(8)))  float  v8f;

#define LN_EPS 1e-5f

// ---------------------------------------------------------------------------
// Generic zero fill
// ---------------------------------------------------------------------------
__global__ void zero_kernel(float* p, long long n) {
    long long i = (long long)blockIdx.x * blockDim.x + threadIdx.x;
    long long stride = (long long)gridDim.x * blockDim.x;
    for (; i < n; i += stride) p[i] = 0.0f;
}

// ---------------------------------------------------------------------------
// Pack fused weights [K=288 x NC] (Ws rows 0..127 | Wm rows 0..127 | Wm rows
// 128..143 | zero pad) into bf16 B-fragment layout for v_wmma_f32_16x16x32_bf16.
// Flat order: tile(kt,nt) -> lane -> 16 contiguous bf16 per lane (32B).
// k(e,lane) = kt*32 + (e<8?0:16) + (lane>=16?8:0) + (e&7);  n = nt*16 + lane%16
// ---------------------------------------------------------------------------
__device__ __forceinline__ void pack_one(int f, int NT, int NC,
                                         const float* Ws, const float* Wm,
                                         unsigned short* pw) {
    int e    = f & 15;
    int lane = (f >> 4) & 31;
    int nt   = (f >> 9) % NT;
    int kt   = f / (512 * NT);
    int k = kt * 32 + ((e < 8) ? 0 : 16) + ((lane >= 16) ? 8 : 0) + (e & 7);
    int n = nt * 16 + (lane & 15);
    float v = 0.0f;
    if (k < 128)       v = Ws[k * NC + n];
    else if (k < 256)  v = Wm[(k - 128) * NC + n];
    else if (k < 272)  v = Wm[(k - 256 + 128) * NC + n];
    __bf16 b = (__bf16)v;
    ((__bf16*)pw)[f] = b;
}

__global__ void pack_weights_kernel(const float* Ws0, const float* Wm0,
                                    const float* Ws1, const float* Wm1,
                                    unsigned short* pw0, unsigned short* pw1) {
    const int T0 = 9 * 8 * 512;   // layer0: 9 k-tiles, 8 n-tiles
    const int T1 = 9 * 4 * 512;   // layer1: 9 k-tiles, 4 n-tiles
    int idx = blockIdx.x * blockDim.x + threadIdx.x;
    if (idx < T0)            pack_one(idx,      8, 128, Ws0, Wm0, pw0);
    else if (idx < T0 + T1)  pack_one(idx - T0, 4,  64, Ws1, Wm1, pw1);
}

// ---------------------------------------------------------------------------
// Edge scatter: agg[tgt] += feat[src] (128 f32), optional agg_ef[tgt] += ef[e]
// (16 f32) and deg[tgt] += 1.  One wave (32 lanes) per edge; float4 per lane.
// All targets are L2-resident (25.6MB << 192MB).
// ---------------------------------------------------------------------------
__global__ __launch_bounds__(256) void edge_scatter_kernel(
    const int* __restrict__ src, const int* __restrict__ tgt,
    const float* __restrict__ feat, const float* __restrict__ ef,
    float* agg, float* agg_ef, float* deg, int E) {
    int e    = blockIdx.x * 8 + (threadIdx.x >> 5);
    int lane = threadIdx.x & 31;
    if (e >= E) return;
    int s = src[e], t = tgt[e];
    float4 v = ((const float4*)(feat + (size_t)s * 128))[lane];
    float* ap = agg + (size_t)t * 128 + lane * 4;
    atomicAdd(ap + 0, v.x);
    atomicAdd(ap + 1, v.y);
    atomicAdd(ap + 2, v.z);
    atomicAdd(ap + 3, v.w);
    if (agg_ef != nullptr && lane < 16)
        atomicAdd(agg_ef + (size_t)t * 16 + lane, ef[(size_t)e * 16 + lane]);
    if (deg != nullptr && lane == 0)
        atomicAdd(deg + t, 1.0f);
}

// ---------------------------------------------------------------------------
// A-fragment assembly: lane holds row m = m0 + lane%16.
// k(e) = (e<8?0:16) + basek + e&7, basek = (lane>=16)?8:0.
// fp32 source, scaled, converted to bf16.
// ---------------------------------------------------------------------------
__device__ __forceinline__ v16bf load_frag_f32(const float* p, float s) {
    const float4* q0 = (const float4*)p;          // k .. k+7
    const float4* q1 = (const float4*)(p + 16);   // k+16 .. k+23
    float4 a0 = q0[0], a1 = q0[1], b0 = q1[0], b1 = q1[1];
    v16bf r;
    r[0] = (__bf16)(a0.x * s); r[1] = (__bf16)(a0.y * s);
    r[2] = (__bf16)(a0.z * s); r[3] = (__bf16)(a0.w * s);
    r[4] = (__bf16)(a1.x * s); r[5] = (__bf16)(a1.y * s);
    r[6] = (__bf16)(a1.z * s); r[7] = (__bf16)(a1.w * s);
    r[8]  = (__bf16)(b0.x * s); r[9]  = (__bf16)(b0.y * s);
    r[10] = (__bf16)(b0.z * s); r[11] = (__bf16)(b0.w * s);
    r[12] = (__bf16)(b1.x * s); r[13] = (__bf16)(b1.y * s);
    r[14] = (__bf16)(b1.z * s); r[15] = (__bf16)(b1.w * s);
    return r;
}

__device__ __forceinline__ v16bf load_frag_ef(const float* p, float s) {
    const float4* q0 = (const float4*)p;  // only K 0..15 valid, rest zero pad
    float4 a0 = q0[0], a1 = q0[1];
    v16bf r;
    r[0] = (__bf16)(a0.x * s); r[1] = (__bf16)(a0.y * s);
    r[2] = (__bf16)(a0.z * s); r[3] = (__bf16)(a0.w * s);
    r[4] = (__bf16)(a1.x * s); r[5] = (__bf16)(a1.y * s);
    r[6] = (__bf16)(a1.z * s); r[7] = (__bf16)(a1.w * s);
    __bf16 z = (__bf16)0.0f;
    r[8] = z; r[9] = z; r[10] = z; r[11] = z;
    r[12] = z; r[13] = z; r[14] = z; r[15] = z;
    return r;
}

// ---------------------------------------------------------------------------
// Node update: out = LN( self@Ws + bs + (agg@WmT + aggEf@WmB)/max(deg,1)
//                        + (deg>0)*bm ) [+ReLU]
// One wave per 16-row strip; NT accumulators cover all NC = NT*16 columns.
// Weights pre-packed as B fragments; K = 9 tiles of 32 (self 128 | agg 128 |
// ef 16 + pad).  Epilogue fuses bias + LayerNorm via __shfl_xor within the
// 16-lane half matching the C-tile layout (VGPR r <-> row r / r+8).
// ---------------------------------------------------------------------------
template<int NT>
__global__ __launch_bounds__(256) void node_layer_kernel(
    const float* __restrict__ selfF,  // [N,128]
    const float* __restrict__ aggF,   // [N,128]
    const float* __restrict__ aggEf,  // [N,16]
    const float* __restrict__ deg,    // [N]
    const unsigned short* __restrict__ Bpack,
    const float* __restrict__ bs, const float* __restrict__ bm,
    const float* __restrict__ gamma, const float* __restrict__ beta,
    float* __restrict__ out, int N, int relu) {
    const int NC = NT * 16;
    int wave = blockIdx.x * 8 + (threadIdx.x >> 5);
    int m0 = wave * 16;
    if (m0 >= N) return;                 // whole wave exits together (EXEC all-1 for WMMA)
    int lane  = threadIdx.x & 31;
    int mA    = m0 + (lane & 15);
    int basek = (lane >= 16) ? 8 : 0;
    float sA  = 1.0f / fmaxf(deg[mA], 1.0f);

    const v16bf* B = (const v16bf*)Bpack;
    const float* selfRow = selfF + (size_t)mA * 128 + basek;
    const float* aggRow  = aggF  + (size_t)mA * 128 + basek;
    const float* efRow   = aggEf + (size_t)mA * 16  + basek;

    v8f acc[NT] = {};
#pragma unroll
    for (int kt = 0; kt < 9; ++kt) {
        v16bf a;
        if (kt < 4)       a = load_frag_f32(selfRow + kt * 32, 1.0f);
        else if (kt < 8)  a = load_frag_f32(aggRow + (kt - 4) * 32, sA);
        else              a = load_frag_ef(efRow, sA);
#pragma unroll
        for (int nt = 0; nt < NT; ++nt) {
            v16bf b = B[(kt * NT + nt) * 32 + lane];
            acc[nt] = __builtin_amdgcn_wmma_f32_16x16x32_bf16(
                false, a, false, b, (short)0, acc[nt], false, false);
        }
    }

    // ---- epilogue: bias + LayerNorm (+ReLU) -------------------------------
    int nloc = lane & 15;
    int half8 = (lane >= 16) ? 8 : 0;
    float bsv[NT], bmv[NT], gv[NT], bev[NT];
#pragma unroll
    for (int nt = 0; nt < NT; ++nt) {
        int n = nt * 16 + nloc;
        bsv[nt] = bs[n]; bmv[nt] = bm[n]; gv[nt] = gamma[n]; bev[nt] = beta[n];
    }
#pragma unroll
    for (int r = 0; r < 8; ++r) {
        int m = m0 + r + half8;                 // row for accumulator element r
        float dflag = (deg[m] > 0.0f) ? 1.0f : 0.0f;
        float v[NT], s = 0.0f, s2 = 0.0f;
#pragma unroll
        for (int nt = 0; nt < NT; ++nt) {
            float t = acc[nt][r] + bsv[nt] + dflag * bmv[nt];
            v[nt] = t; s += t; s2 += t * t;
        }
        // reduce across the 16 lanes of this half (xor masks 1,2,4,8 stay in-half)
#pragma unroll
        for (int mask = 1; mask < 16; mask <<= 1) {
            s  += __shfl_xor(s,  mask, 32);
            s2 += __shfl_xor(s2, mask, 32);
        }
        float mu  = s / (float)NC;
        float var = s2 / (float)NC - mu * mu;
        float inv = rsqrtf(var + LN_EPS);
#pragma unroll
        for (int nt = 0; nt < NT; ++nt) {
            float o = (v[nt] - mu) * inv * gv[nt] + bev[nt];
            if (relu) o = fmaxf(o, 0.0f);
            out[(size_t)m * NC + nt * 16 + nloc] = o;
        }
    }
}

// ---------------------------------------------------------------------------
// Graph mean pooling
// ---------------------------------------------------------------------------
__global__ void pool_accum_kernel(const float* __restrict__ node_emb,
                                  const int* __restrict__ batch,
                                  float* gacc, float* counts, long long total) {
    long long idx = (long long)blockIdx.x * blockDim.x + threadIdx.x;
    if (idx >= total) return;
    int node = (int)(idx >> 6);
    int c    = (int)(idx & 63);
    int g = batch[node];
    atomicAdd(gacc + (size_t)g * 64 + c, node_emb[idx]);
    if (c == 0) atomicAdd(counts + g, 1.0f);
}

__global__ void pool_final_kernel(float* gacc, const float* __restrict__ counts,
                                  int total) {
    int idx = blockIdx.x * blockDim.x + threadIdx.x;
    if (idx >= total) return;
    gacc[idx] /= fmaxf(counts[idx >> 6], 1.0f);
}

// ---------------------------------------------------------------------------
extern "C" void kernel_launch(void* const* d_in, const int* in_sizes, int n_in,
                              void* d_out, int out_size, void* d_ws, size_t ws_size,
                              hipStream_t stream) {
    // setup_inputs() order:
    //  0:x 1:edge_features 2:Ws0 3:bs0 4:Wm0 5:bm0 6:g0 7:be0
    //  8:Ws1 9:bs1 10:Wm1 11:bm1 12:g1 13:be1 14:edge_index 15:batch 16:num_graphs
    const float* x    = (const float*)d_in[0];
    const float* ef   = (const float*)d_in[1];
    const float* Ws0  = (const float*)d_in[2];
    const float* bs0  = (const float*)d_in[3];
    const float* Wm0  = (const float*)d_in[4];
    const float* bm0  = (const float*)d_in[5];
    const float* g0   = (const float*)d_in[6];
    const float* be0  = (const float*)d_in[7];
    const float* Ws1  = (const float*)d_in[8];
    const float* bs1  = (const float*)d_in[9];
    const float* Wm1  = (const float*)d_in[10];
    const float* bm1  = (const float*)d_in[11];
    const float* g1   = (const float*)d_in[12];
    const float* be1  = (const float*)d_in[13];
    const int*   eidx  = (const int*)d_in[14];
    const int*   batch = (const int*)d_in[15];

    int N = in_sizes[0] / 128;
    int E = in_sizes[14] / 2;
    int G = (out_size - N * 64) / 64;
    const int* src = eidx;
    const int* tgt = eidx + E;

    // workspace layout
    float* agg    = (float*)d_ws;                 // [N,128] (agg_x, reused as agg_h)
    float* aggEf  = agg   + (size_t)N * 128;      // [N,16]
    float* deg    = aggEf + (size_t)N * 16;       // [N]
    float* counts = deg + N;                      // [G]
    float* h      = counts + G;                   // [N,128]
    uintptr_t pwp = (uintptr_t)(h + (size_t)N * 128);
    pwp = (pwp + 63) & ~(uintptr_t)63;
    unsigned short* pw0 = (unsigned short*)pwp;   // 9*8*512 bf16
    unsigned short* pw1 = pw0 + 9 * 8 * 512;      // 9*4*512 bf16

    float* node_emb = (float*)d_out;              // [N,64]
    float* gacc     = node_emb + (size_t)N * 64;  // [G,64]

    // 1. pack fused weights into WMMA B-fragment layout
    {
        int tot = 9 * 8 * 512 + 9 * 4 * 512;
        pack_weights_kernel<<<(tot + 255) / 256, 256, 0, stream>>>(
            Ws0, Wm0, Ws1, Wm1, pw0, pw1);
    }
    // 2. zero agg_x | agg_ef | deg | counts (contiguous) and graph accumulator
    {
        long long n1 = (long long)N * 144 + N + G;
        zero_kernel<<<2048, 256, 0, stream>>>(agg, n1);
        zero_kernel<<<(G * 64 + 255) / 256, 256, 0, stream>>>(gacc, (long long)G * 64);
    }
    // 3. layer-0 edge scatter: agg_x += x[src], agg_ef += ef, deg += 1
    edge_scatter_kernel<<<(E + 7) / 8, 256, 0, stream>>>(
        src, tgt, x, ef, agg, aggEf, deg, E);
    // 4. layer-0 node update -> h  (H=128, relu)
    {
        int waves = N / 16;
        node_layer_kernel<8><<<(waves + 7) / 8, 256, 0, stream>>>(
            x, agg, aggEf, deg, pw0, bs0, bm0, g0, be0, h, N, 1);
    }
    // 5. re-zero agg buffer for layer 1
    zero_kernel<<<2048, 256, 0, stream>>>(agg, (long long)N * 128);
    // 6. layer-1 edge scatter: agg_h += h[src]
    edge_scatter_kernel<<<(E + 7) / 8, 256, 0, stream>>>(
        src, tgt, h, nullptr, agg, nullptr, nullptr, E);
    // 7. layer-1 node update -> node_emb (L=64, no relu)
    {
        int waves = N / 16;
        node_layer_kernel<4><<<(waves + 7) / 8, 256, 0, stream>>>(
            h, agg, aggEf, deg, pw1, bs1, bm1, g1, be1, node_emb, N, 0);
    }
    // 8. graph mean pooling
    {
        long long tot = (long long)N * 64;
        pool_accum_kernel<<<(unsigned)((tot + 255) / 256), 256, 0, stream>>>(
            node_emb, batch, gacc, counts, tot);
        pool_final_kernel<<<(G * 64 + 255) / 256, 256, 0, stream>>>(gacc, counts, G * 64);
    }
}
